// GroupedMLP_44702019617155
// MI455X (gfx1250) — matrix operand order repaired and checked by reading, not compiled, not run on hardware
//
#include <hip/hip_runtime.h>
#include <hip/hip_bf16.h>
#include <stdint.h>

// ---------------------------------------------------------------------------
// Grouped SwiGLU MLP for MI455X (gfx1250), bf16 WMMA path, double-buffered.
//   Kernel 1: hidden = silu(X @ Wg) * (X @ Wu)   (bf16 result into d_ws)
//   Kernel 2: out    = hidden @ Wd               (fp32 result into d_out)
// Shapes: X (4096*8, 1024) with token row = b*8 + g; Wg/Wu (8,1024,2048);
//         Wd (8,2048,1024). Workspace needs 8*4096*2048*2 = 134.2 MB.
// ---------------------------------------------------------------------------

#define BTOK   4096
#define NGRP   8
#define DIN    1024
#define DHID   2048
#define DOUT   1024

#define BM     128       // rows per block tile
#define BN     128       // cols per block tile
#define KB     32        // K step == WMMA K
#define LDSK   40        // bf16 elems per LDS row (32 used + 8 pad -> 80B stride)

typedef __attribute__((ext_vector_type(16))) __bf16 v16bf;
typedef __attribute__((ext_vector_type(8)))  __bf16 bf16x8;
typedef __attribute__((ext_vector_type(2)))  __bf16 bf16x2;
typedef __attribute__((ext_vector_type(8)))  float  v8f;
typedef __attribute__((ext_vector_type(4)))  float  f32x4;
typedef __attribute__((ext_vector_type(2)))  float  f32x2;
typedef __attribute__((ext_vector_type(2)))  unsigned u32x2;
typedef __attribute__((ext_vector_type(4)))  int    i32x4;

union Frag { v16bf v; bf16x8 h[2]; };

// Native packed fp32->bf16 (compiler lowers to v_cvt_pk_bf16_f32-class ops).
__device__ __forceinline__ unsigned cvt2(float lo, float hi) {
    f32x2 f; f.x = lo; f.y = hi;
    union { bf16x2 b; unsigned u; } o;
    o.b = __builtin_convertvector(f, bf16x2);
    return o.u;
}

// ---- CDNA5 async global->LDS copy (ASYNCcnt path), guarded ----------------
__device__ __forceinline__ void async_copy16(const void* gsrc, void* ldst) {
#if __has_builtin(__builtin_amdgcn_global_load_async_to_lds_b128)
    __builtin_amdgcn_global_load_async_to_lds_b128(
        (__attribute__((address_space(1))) i32x4*)gsrc,
        (__attribute__((address_space(3))) i32x4*)ldst, 0, 0);
#else
    *(bf16x8*)ldst = *(const bf16x8*)gsrc;   // synchronous fallback
#endif
}
__device__ __forceinline__ void async_wait0() {
#if __has_builtin(__builtin_amdgcn_s_wait_asynccnt)
    __builtin_amdgcn_s_wait_asynccnt(0);
#else
    asm volatile("s_wait_asynccnt 0" ::: "memory");
#endif
}

// A fragment: 16x32 bf16, LDS layout [m][k] (k contiguous, LDSK stride).
// Lane l (row M = rowbase + l%16, half = l/16):
//   elems 0-7  = K[8*half .. 8*half+7], elems 8-15 = K[16+8*half ..].
__device__ __forceinline__ v16bf load_a_frag(const __bf16* lds, int rowbase) {
    const int lane = threadIdx.x & 31;
    const int half = lane >> 4, r = lane & 15;
    const char* base = (const char*)(lds + (rowbase + r) * LDSK);
    Frag f;
    f.h[0] = *(const bf16x8*)(base + 16 * half);
    f.h[1] = *(const bf16x8*)(base + 32 + 16 * half);
    return f.v;
}

// B fragment: 32x16 bf16, LDS layout [n][k] (transposed weight, k contiguous).
// Lane l (col N = nbase + l%16): elems 0-7 = K[8*half ..], elems 8-15 = K[16+8*half ..].
__device__ __forceinline__ v16bf load_b_frag(const __bf16* lds, int nbase) {
    const int lane = threadIdx.x & 31;
    const int half = lane >> 4, r = lane & 15;
    const char* base = (const char*)(lds + (nbase + r) * LDSK);
    Frag f;
    f.h[0] = *(const bf16x8*)(base + 16 * half);
    f.h[1] = *(const bf16x8*)(base + 32 + 16 * half);
    return f.v;
}

// Stage a KBxBN fp32 weight tile (row-major [k][n], row stride ldW) into LDS
// transposed/bf16 as [n][k]. Each thread: 2 adjacent k rows x 8 n values.
__device__ __forceinline__ void stage_w(const float* __restrict__ W, int ldW,
                                        int k0, int n0, __bf16* lds, bool pf) {
    const int hi_ = (threadIdx.x & 15) * 8;       // n within tile: 0..120
    const int kp  = (threadIdx.x >> 4) * 2;       // k pair base:   0..30
    const float* w0 = W + (size_t)(k0 + kp) * ldW + n0 + hi_;
    if (pf) __builtin_prefetch(w0 + (size_t)2 * KB * ldW, 0, 1);  // k0+2*KB
    f32x4 a0 = *(const f32x4*)(w0);
    f32x4 a1 = *(const f32x4*)(w0 + 4);
    f32x4 b0 = *(const f32x4*)(w0 + ldW);
    f32x4 b1 = *(const f32x4*)(w0 + ldW + 4);
    float ka[8] = {a0.x, a0.y, a0.z, a0.w, a1.x, a1.y, a1.z, a1.w};
    float kb[8] = {b0.x, b0.y, b0.z, b0.w, b1.x, b1.y, b1.z, b1.w};
#pragma unroll
    for (int j = 0; j < 8; ++j) {
        *(unsigned*)(&lds[(hi_ + j) * LDSK + kp]) = cvt2(ka[j], kb[j]);
    }
}

// Stage 128x32 fp32 X tile (rows of group g strided by NGRP) as bf16 [m][k].
__device__ __forceinline__ void stage_x(const float* __restrict__ X, int g,
                                        int m0, int k0, __bf16* lds) {
    const int kk = (threadIdx.x & 7) * 4;
    const int mb = threadIdx.x >> 3;
#pragma unroll
    for (int i = 0; i < 4; ++i) {
        const int m = mb + 32 * i;
        const float* src = X + ((size_t)(m0 + m) * NGRP + g) * DIN + k0 + kk;
        f32x4 v = *(const f32x4*)src;
        u32x2 p;
        p.x = cvt2(v.x, v.y);
        p.y = cvt2(v.z, v.w);
        *(u32x2*)(&lds[m * LDSK + kk]) = p;
    }
}

// Stage 128x32 bf16 hidden tile via async global->LDS (no conversion needed).
__device__ __forceinline__ void stage_h(const __bf16* __restrict__ hidden,
                                        int g, int m0, int k0, __bf16* lds) {
#pragma unroll
    for (int i = 0; i < 2; ++i) {
        const int c = threadIdx.x + 256 * i;   // 512 chunks of 8 bf16 (16B)
        const int m = c >> 2, q = c & 3;
        const __bf16* src =
            hidden + ((size_t)g * BTOK + m0 + m) * DHID + k0 + q * 8;
        async_copy16(src, &lds[m * LDSK + q * 8]);
    }
}

// ---------------------------------------------------------------------------
// Kernel 1: gate/up GEMMs + fused SwiGLU -> bf16 hidden
// grid: (DHID/BN=16, BTOK/BM=32, NGRP=8), block: 256 (8 wave32)
// ---------------------------------------------------------------------------
__global__ __launch_bounds__(256)
void swiglu_gateup_kernel(const float* __restrict__ X,
                          const float* __restrict__ Wg,
                          const float* __restrict__ Wu,
                          __bf16* __restrict__ hidden) {
    __shared__ __bf16 lds_a [2][BM * LDSK];
    __shared__ __bf16 lds_bg[2][BN * LDSK];
    __shared__ __bf16 lds_bu[2][BN * LDSK];

    const int g  = blockIdx.z;
    const int m0 = blockIdx.y * BM;
    const int h0 = blockIdx.x * BN;

    const float* wg = Wg + (size_t)g * DIN * DHID;
    const float* wu = Wu + (size_t)g * DIN * DHID;

    const int wave = threadIdx.x >> 5;
    const int wm = wave >> 2;     // 0..1  (64 rows each)
    const int wn = wave & 3;      // 0..3  (32 cols each)

    v8f accg[4][2] = {};
    v8f accu[4][2] = {};

    // prologue: fill buffer 0
    stage_x(X, g, m0, 0, lds_a[0]);
    stage_w(wg, DHID, 0, h0, lds_bg[0], true);
    stage_w(wu, DHID, 0, h0, lds_bu[0], true);

    int p = 0;
    for (int k0 = 0; k0 < DIN; k0 += KB) {
        __syncthreads();                      // publish buffer p
        const int kn = k0 + KB;
        if (kn < DIN) {                       // stage next tile into p^1
            stage_x(X, g, m0, kn, lds_a[p ^ 1]);
            stage_w(wg, DHID, kn, h0, lds_bg[p ^ 1], kn + KB < DIN);
            stage_w(wu, DHID, kn, h0, lds_bu[p ^ 1], kn + KB < DIN);
        }

        v16bf af[4], bg[2], bu[2];
#pragma unroll
        for (int tm = 0; tm < 4; ++tm)
            af[tm] = load_a_frag(lds_a[p], wm * 64 + tm * 16);
#pragma unroll
        for (int tn = 0; tn < 2; ++tn) {
            bg[tn] = load_b_frag(lds_bg[p], wn * 32 + tn * 16);
            bu[tn] = load_b_frag(lds_bu[p], wn * 32 + tn * 16);
        }
#pragma unroll
        for (int tm = 0; tm < 4; ++tm)
#pragma unroll
            for (int tn = 0; tn < 2; ++tn) {
                accg[tm][tn] = __builtin_amdgcn_wmma_f32_16x16x32_bf16(
                    false, af[tm], false, bg[tn], (short)0, accg[tm][tn], false, false);
                accu[tm][tn] = __builtin_amdgcn_wmma_f32_16x16x32_bf16(
                    false, af[tm], false, bu[tn], (short)0, accu[tm][tn], false, false);
            }
        p ^= 1;
    }

    // ---- fused SwiGLU epilogue, bf16 store to workspace ----
    const int lane = threadIdx.x & 31;
    const int half = lane >> 4, r = lane & 15;
#pragma unroll
    for (int tm = 0; tm < 4; ++tm)
#pragma unroll
        for (int tn = 0; tn < 2; ++tn)
#pragma unroll
            for (int v = 0; v < 8; ++v) {
                float gt = accg[tm][tn][v];
                float uv = accu[tm][tn][v];
                float s  = gt / (1.0f + __expf(-gt));   // silu
                int row = m0 + wm * 64 + tm * 16 + v + 8 * half;
                int col = h0 + wn * 32 + tn * 16 + r;
                hidden[((size_t)g * BTOK + row) * DHID + col] = (__bf16)(s * uv);
            }
}

// ---------------------------------------------------------------------------
// Kernel 2: out = hidden(bf16) @ Wd ; A tile staged via async global->LDS
// grid: (DOUT/BN=8, BTOK/BM=32, NGRP=8), block: 256 (8 wave32)
// ---------------------------------------------------------------------------
__global__ __launch_bounds__(256)
void down_proj_kernel(const __bf16* __restrict__ hidden,
                      const float* __restrict__ Wd,
                      float* __restrict__ out) {
    __shared__ __bf16 lds_a[2][BM * LDSK];
    __shared__ __bf16 lds_b[2][BN * LDSK];

    const int g  = blockIdx.z;
    const int m0 = blockIdx.y * BM;
    const int n0 = blockIdx.x * BN;

    const float* wd = Wd + (size_t)g * DHID * DOUT;

    const int wave = threadIdx.x >> 5;
    const int wm = wave >> 2;
    const int wn = wave & 3;

    v8f acc[4][2] = {};

    // prologue: fill buffer 0 (async A copy + converted B tile)
    stage_h(hidden, g, m0, 0, lds_a[0]);
    stage_w(wd, DOUT, 0, n0, lds_b[0], true);
    async_wait0();                            // our async LDS writes landed

    int p = 0;
    for (int k0 = 0; k0 < DHID; k0 += KB) {
        __syncthreads();                      // publish buffer p
        const int kn = k0 + KB;
        const bool nxt = kn < DHID;
        if (nxt) {                            // issue next tile into p^1
            stage_h(hidden, g, m0, kn, lds_a[p ^ 1]);
            stage_w(wd, DOUT, kn, n0, lds_b[p ^ 1], kn + KB < DHID);
        }

        v16bf af[4], bf[2];
#pragma unroll
        for (int tm = 0; tm < 4; ++tm)
            af[tm] = load_a_frag(lds_a[p], wm * 64 + tm * 16);
#pragma unroll
        for (int tn = 0; tn < 2; ++tn)
            bf[tn] = load_b_frag(lds_b[p], wn * 32 + tn * 16);
#pragma unroll
        for (int tm = 0; tm < 4; ++tm)
#pragma unroll
            for (int tn = 0; tn < 2; ++tn)
                acc[tm][tn] = __builtin_amdgcn_wmma_f32_16x16x32_bf16(
                    false, af[tm], false, bf[tn], (short)0, acc[tm][tn], false, false);

        if (nxt) async_wait0();               // drain before publishing barrier
        p ^= 1;
    }

    // ---- fp32 epilogue; token row in output = b*NGRP + g ----
    const int lane = threadIdx.x & 31;
    const int half = lane >> 4, r = lane & 15;
#pragma unroll
    for (int tm = 0; tm < 4; ++tm)
#pragma unroll
        for (int tn = 0; tn < 2; ++tn)
#pragma unroll
            for (int v = 0; v < 8; ++v) {
                int row = m0 + wm * 64 + tm * 16 + v + 8 * half;   // b index
                int col = n0 + wn * 32 + tn * 16 + r;
                out[((size_t)row * NGRP + g) * DOUT + col] = acc[tm][tn][v];
            }
}

// ---------------------------------------------------------------------------
extern "C" void kernel_launch(void* const* d_in, const int* in_sizes, int n_in,
                              void* d_out, int out_size, void* d_ws, size_t ws_size,
                              hipStream_t stream) {
    const float* x  = (const float*)d_in[0];
    const float* wg = (const float*)d_in[1];
    const float* wu = (const float*)d_in[2];
    const float* wd = (const float*)d_in[3];
    (void)in_sizes; (void)n_in; (void)out_size; (void)ws_size;

    __bf16* hidden = (__bf16*)d_ws;   // 8*4096*2048 bf16 = 134.2 MB

    dim3 blk(256, 1, 1);
    dim3 grd1(DHID / BN, BTOK / BM, NGRP);   // (16, 32, 8)
    dim3 grd2(DOUT / BN, BTOK / BM, NGRP);   // ( 8, 32, 8)

    swiglu_gateup_kernel<<<grd1, blk, 0, stream>>>(x, wg, wu, hidden);
    down_proj_kernel   <<<grd2, blk, 0, stream>>>(hidden, wd, (float*)d_out);
}
